// PVMFormer_81346680586880
// MI455X (gfx1250) — compile-verified, matrix operand order checked by reading, not run on hardware
//
#include <hip/hip_runtime.h>
#include <hip/hip_bf16.h>
#include <math.h>

// ---------------- types / helpers ----------------
typedef __attribute__((ext_vector_type(16))) __bf16          v16bf;
typedef __attribute__((ext_vector_type(8)))  float           v8f;

__device__ __forceinline__ unsigned short f2bf(float f) {
    unsigned int u = __float_as_uint(f);
    unsigned int r = u + 0x7FFFu + ((u >> 16) & 1u);   // RNE
    return (unsigned short)(r >> 16);
}

// Load a 16x32 bf16 A/B fragment from bf16 (u16) row-major source with two
// 16-byte vector loads. base = tile's first row; ld = row stride (elems,
// multiple of 8 -> 16B aligned); k0 = K offset (multiple of 32).
__device__ __forceinline__ v16bf load_frag_bf16(const unsigned short* base, int ld, int k0) {
    int lane = threadIdx.x & 31;
    int rm   = lane & 15;
    int kb   = (lane >> 4) * 8;
    const unsigned short* p = base + (size_t)rm * ld + k0 + kb;
    union { uint4 q[2]; v16bf b; } f;
    f.q[0] = *(const uint4*)(p);        // K = k0+kb   .. +7
    f.q[1] = *(const uint4*)(p + 16);   // K = k0+kb+16.. +23
    return f.b;
}

// 1 A-tile x 4 W-tiles inner product over K.
template<int KDIM>
__device__ __forceinline__ void mma_loop4(const unsigned short* __restrict__ Ab,
                                          const unsigned short* __restrict__ Wb,
                                          v8f* acc) {
    for (int k0 = 0; k0 < KDIM; k0 += 32) {
        v16bf a = load_frag_bf16(Ab, KDIM, k0);
#pragma unroll
        for (int t = 0; t < 4; ++t) {
            v16bf b = load_frag_bf16(Wb + (size_t)(t * 16) * KDIM, KDIM, k0);
            acc[t] = __builtin_amdgcn_wmma_f32_16x16x32_bf16(false, a, false, b,
                                                             (short)0, acc[t], false, false);
        }
    }
}

__device__ __forceinline__ float silu_f(float v) { return v / (1.f + __expf(-v)); }

// ---------------- constants ----------------
#define Bb   4
#define Cc   256
#define LL   4096
#define DM   64
#define DI   128
#define DS   16
#define DR   4
#define NSEQ 16                     // 4*B
#define NCH  16                     // scan chunks per sequence
#define CL   256                    // LL / NCH
#define MROWS ((size_t)NSEQ * LL)   // 65536
#define PROWS ((size_t)Bb * LL)     // 16384

// ---------------- kernel 0: generic f32 -> bf16 ----------------
__global__ void cvt_bf16_kernel(const float* __restrict__ in,
                                unsigned short* __restrict__ out, int n) {
    int i = blockIdx.x * 256 + threadIdx.x;
    if (i < n) out[i] = f2bf(in[i]);
}

// ---------------- kernel 1: LN1 + chunk scatter (fp32 + bf16) ----------------
__global__ void ln1_chunks_kernel(const float* __restrict__ x,
                                  const float* __restrict__ lnw,
                                  const float* __restrict__ lnb,
                                  float* __restrict__ chunks,
                                  unsigned short* __restrict__ chunks_bf) {
    int row = blockIdx.x;          // b*L + l
    int b = row >> 12, l = row & (LL - 1);
    int c = threadIdx.x;           // 0..255
    __shared__ float s1[Cc], s2[Cc];
    float v = x[((size_t)b * Cc + c) * LL + l];
    s1[c] = v; s2[c] = v * v;
    __syncthreads();
    for (int st = Cc / 2; st > 0; st >>= 1) {
        if (c < st) { s1[c] += s1[c + st]; s2[c] += s2[c + st]; }
        __syncthreads();
    }
    float mu = s1[0] * (1.f / Cc);
    float var = s2[0] * (1.f / Cc) - mu * mu;
    float rstd = rsqrtf(var + 1e-5f);
    float y = (v - mu) * rstd * lnw[c] + lnb[c];
    int q = c >> 6, m = c & 63;
    int s = q * Bb + b;
    size_t idx = ((size_t)s * LL + l) * DM + m;
    chunks[idx] = y;
    chunks_bf[idx] = f2bf(y);
}

// ---------------- kernel 2: xz = chunks @ in_w.T (WMMA, K=64) ----------------
__global__ void gemm_in_kernel(const unsigned short* __restrict__ A,  // 65536 x 64 bf16
                               const unsigned short* __restrict__ W,  // 256 x 64 bf16
                               float* __restrict__ Out) {             // 65536 x 256
    int mt  = blockIdx.x * 4 + (threadIdx.x >> 5);
    int nt4 = blockIdx.y;
    const unsigned short* Ab = A + (size_t)mt * 16 * DM;
    const unsigned short* Wb = W + (size_t)nt4 * 64 * DM;
    v8f acc[4] = {};
    mma_loop4<DM>(Ab, Wb, acc);
    int lane = threadIdx.x & 31;
    int mb = (lane >> 4) * 8;
#pragma unroll
    for (int t = 0; t < 4; ++t) {
        int n = nt4 * 64 + t * 16 + (lane & 15);
#pragma unroll
        for (int v = 0; v < 8; ++v) {
            size_t m = (size_t)mt * 16 + mb + v;
            Out[m * 256 + n] = acc[t][v];
        }
    }
}

// ---------------- kernel 3: depthwise causal conv + SiLU ----------------
__global__ void conv_silu_kernel(const float* __restrict__ xz,   // 65536 x 256
                                 const float* __restrict__ cw,   // 128 x 1 x 4
                                 const float* __restrict__ cb,   // 128
                                 float* __restrict__ xc) {       // 65536 x 128
    size_t idx = (size_t)blockIdx.x * blockDim.x + threadIdx.x;
    if (idx >= MROWS * DI) return;
    int d = (int)(idx & (DI - 1));
    size_t r = idx >> 7;                // s*L + l
    int l = (int)(r & (LL - 1));
    size_t rowb = r - l;
    float acc = cb[d];
#pragma unroll
    for (int k = 0; k < 4; ++k) {
        int ll = l - 3 + k;
        if (ll >= 0) acc += cw[d * 4 + k] * xz[(rowb + ll) * 256 + d];
    }
    xc[idx] = silu_f(acc);
}

// ---------------- kernel 4: x-proj + dt softplus ----------------
__global__ void xproj_dt_kernel(const float* __restrict__ xc,
                                const float* __restrict__ xpw,  // 36 x 128
                                const float* __restrict__ dtw,  // 128 x 4
                                const float* __restrict__ dtb,  // 128
                                float* __restrict__ dt,         // 65536 x 128
                                float* __restrict__ Bs,         // 65536 x 16
                                float* __restrict__ Cs) {       // 65536 x 16
    size_t r = blockIdx.x;
    int t = threadIdx.x;     // 0..127
    __shared__ float row[DI];
    __shared__ float dbl[DR + 2 * DS];
    row[t] = xc[r * DI + t];
    __syncthreads();
    if (t < DR + 2 * DS) {
        float acc = 0.f;
        for (int k = 0; k < DI; ++k) acc += xpw[t * DI + k] * row[k];
        dbl[t] = acc;
    }
    __syncthreads();
    if (t < DS)            Bs[r * DS + t]        = dbl[DR + t];
    else if (t < 2 * DS)   Cs[r * DS + (t - DS)] = dbl[DR + DS + (t - DS)];
    float acc = dtb[t];
#pragma unroll
    for (int j = 0; j < DR; ++j) acc += dtw[t * DR + j] * dbl[j];
    float sp = (acc > 20.f) ? acc : logf(1.f + __expf(acc));
    dt[r * DI + t] = sp;
}

// ---------------- kernel 5a: scan pass 1 — per-chunk transition (P, S) ----------------
// h_out = P * h_in + S  where P = prod(dA), S = chunk-local scan from h=0.
__global__ void scan_chunk1_kernel(const float* __restrict__ dt,
                                   const float* __restrict__ Bs,
                                   const float* __restrict__ xc,
                                   const float* __restrict__ A_log,
                                   float* __restrict__ chP,     // [256][128][16]
                                   float* __restrict__ chS) {   // [256][128][16]
    int blk = blockIdx.x;            // seq*NCH + chunk
    int s = blk >> 4, ch = blk & 15;
    int d = threadIdx.x;             // 0..127
    int lane = d & 31;
    float a[DS], P[DS], S[DS];
#pragma unroll
    for (int j = 0; j < DS; ++j) {
        a[j] = -__expf(A_log[d * DS + j]); P[j] = 1.f; S[j] = 0.f;
    }
    size_t rb = (size_t)s * LL + (size_t)ch * CL;
    for (int l = 0; l < CL; ++l) {
        size_t r = rb + l;
        float bv  = Bs[r * DS + (lane & 15)];   // lanes 0-15 & 16-31 mirror B
        float dtv = dt[r * DI + d];
        float xv  = xc[r * DI + d];
        float dx  = dtv * xv;
#pragma unroll
        for (int j = 0; j < DS; ++j) {
            float dA = __expf(dtv * a[j]);
            float bj = __shfl(bv, j, 32);
            P[j] *= dA;
            S[j] = dA * S[j] + dx * bj;
        }
    }
    size_t base = ((size_t)blk * DI + d) * DS;
#pragma unroll
    for (int j = 0; j < DS; ++j) { chP[base + j] = P[j]; chS[base + j] = S[j]; }
}

// ---------------- kernel 5b: scan pass 2 — prefix over chunks ----------------
__global__ void scan_prefix_kernel(const float* __restrict__ chP,
                                   const float* __restrict__ chS,
                                   float* __restrict__ h0) {    // [256][128][16]
    int idx = blockIdx.x * 256 + threadIdx.x;   // seq*(DI*DS) + d*DS + s : 32768 total
    int seq = idx >> 11;                        // / (128*16)
    int rem = idx & 2047;
    float h = 0.f;
#pragma unroll
    for (int c = 0; c < NCH; ++c) {
        size_t base = ((size_t)(seq * NCH + c)) * (DI * DS) + rem;
        h0[base] = h;
        h = chP[base] * h + chS[base];
    }
}

// ---------------- kernel 5c: scan pass 3 — recompute outputs from exact h_in ----------------
__global__ void scan_chunk2_kernel(const float* __restrict__ dt,
                                   const float* __restrict__ Bs,
                                   const float* __restrict__ Cs,
                                   const float* __restrict__ xc,
                                   const float* __restrict__ xz,     // z = cols 128..255
                                   const float* __restrict__ A_log,
                                   const float* __restrict__ Dp,
                                   const float* __restrict__ h0,
                                   unsigned short* __restrict__ ybuf) { // 65536 x 128 bf16
    int blk = blockIdx.x;            // seq*NCH + chunk
    int s = blk >> 4, ch = blk & 15;
    int d = threadIdx.x;
    int lane = d & 31;
    float a[DS], h[DS];
    size_t hb = ((size_t)blk * DI + d) * DS;
#pragma unroll
    for (int j = 0; j < DS; ++j) { a[j] = -__expf(A_log[d * DS + j]); h[j] = h0[hb + j]; }
    float dpv = Dp[d];
    size_t rb = (size_t)s * LL + (size_t)ch * CL;
    for (int l = 0; l < CL; ++l) {
        size_t r = rb + l;
        // lanes 0-15 carry B[0..15], lanes 16-31 carry C[0..15]; wave-local broadcast
        float bcv = (lane < DS) ? Bs[r * DS + lane] : Cs[r * DS + (lane - DS)];
        float dtv = dt[r * DI + d];
        float xv  = xc[r * DI + d];
        float dx  = dtv * xv;
        float y = 0.f;
#pragma unroll
        for (int j = 0; j < DS; ++j) {
            float dA = __expf(dtv * a[j]);
            float bj = __shfl(bcv, j, 32);
            float cj = __shfl(bcv, DS + j, 32);
            h[j] = dA * h[j] + dx * bj;
            y += h[j] * cj;
        }
        float zv = xz[r * 256 + DI + d];
        ybuf[r * DI + d] = f2bf((y + xv * dpv) * silu_f(zv));
    }
}

// ---------------- kernel 6: ym = ybuf @ out_w.T + skip*chunks -> (B,L,C) ----------------
__global__ void gemm_out_kernel(const unsigned short* __restrict__ A,  // 65536 x 128 bf16
                                const unsigned short* __restrict__ W,  // 64 x 128 bf16
                                const float* __restrict__ chunks,      // 65536 x 64 fp32
                                const float* __restrict__ skip,        // 1
                                float* __restrict__ ym_blc) {          // 16384 x 256
    int mt = blockIdx.x * 4 + (threadIdx.x >> 5);
    const unsigned short* Ab = A + (size_t)mt * 16 * DI;
    v8f acc[4] = {};
    mma_loop4<DI>(Ab, W, acc);       // N = 64 -> all 4 tiles
    int lane = threadIdx.x & 31;
    int mb = (lane >> 4) * 8;
    float ss = skip[0];
#pragma unroll
    for (int t = 0; t < 4; ++t) {
        int n = t * 16 + (lane & 15);   // 0..63
#pragma unroll
        for (int v = 0; v < 8; ++v) {
            size_t m = (size_t)mt * 16 + mb + v;     // r = s*L + l
            int l = (int)(m & (LL - 1));
            int sq = (int)(m >> 12);
            int q = sq >> 2, b = sq & 3;
            float val = acc[t][v] + ss * chunks[m * DM + n];
            ym_blc[((size_t)b * LL + l) * Cc + q * DM + n] = val;
        }
    }
}

// ---------------- kernel 7: LN2 -> bf16 ----------------
__global__ void ln2_kernel(const float* __restrict__ in,
                           const float* __restrict__ lnw,
                           const float* __restrict__ lnb,
                           unsigned short* __restrict__ out) {
    size_t row = blockIdx.x;
    int c = threadIdx.x;
    __shared__ float s1[Cc], s2[Cc];
    float v = in[row * Cc + c];
    s1[c] = v; s2[c] = v * v;
    __syncthreads();
    for (int st = Cc / 2; st > 0; st >>= 1) {
        if (c < st) { s1[c] += s1[c + st]; s2[c] += s2[c + st]; }
        __syncthreads();
    }
    float mu = s1[0] * (1.f / Cc);
    float var = s2[0] * (1.f / Cc) - mu * mu;
    float rstd = rsqrtf(var + 1e-5f);
    out[row * Cc + c] = f2bf((v - mu) * rstd * lnw[c] + lnb[c]);
}

// ---------------- kernel 8: proj GEMM + bias + x residual ----------------
__global__ void gemm_proj_kernel(const unsigned short* __restrict__ A,  // 16384 x 256 bf16
                                 const unsigned short* __restrict__ W,  // 256 x 256 bf16
                                 const float* __restrict__ pb,          // 256
                                 const float* __restrict__ x,           // B,C,H,W
                                 float* __restrict__ out_blc,           // 16384 x 256 fp32
                                 unsigned short* __restrict__ out_bf) { // 16384 x 256 bf16
    int mt  = blockIdx.x * 4 + (threadIdx.x >> 5);
    int nt4 = blockIdx.y;
    const unsigned short* Ab = A + (size_t)mt * 16 * Cc;
    const unsigned short* Wb = W + (size_t)nt4 * 64 * Cc;
    v8f acc[4] = {};
    mma_loop4<Cc>(Ab, Wb, acc);
    int lane = threadIdx.x & 31;
    int mb = (lane >> 4) * 8;
#pragma unroll
    for (int t = 0; t < 4; ++t) {
        int n = nt4 * 64 + t * 16 + (lane & 15);
#pragma unroll
        for (int v = 0; v < 8; ++v) {
            size_t m = (size_t)mt * 16 + mb + v;   // r = b*L + l
            int b = (int)(m >> 12), l = (int)(m & (LL - 1));
            float val = acc[t][v] + pb[n] + x[((size_t)b * Cc + n) * LL + l];
            out_blc[m * Cc + n] = val;
            out_bf[m * Cc + n]  = f2bf(val);
        }
    }
}

// ---------------- kernel 9: fc1 GEMM + GELU -> bf16 ----------------
__global__ void gemm_fc1_kernel(const unsigned short* __restrict__ A,  // 16384 x 256 bf16
                                const unsigned short* __restrict__ W,  // 1024 x 256 bf16
                                const float* __restrict__ fb,          // 1024
                                unsigned short* __restrict__ H) {      // 16384 x 1024 bf16
    int mt  = blockIdx.x * 4 + (threadIdx.x >> 5);
    int nt4 = blockIdx.y;
    const unsigned short* Ab = A + (size_t)mt * 16 * Cc;
    const unsigned short* Wb = W + (size_t)nt4 * 64 * Cc;
    v8f acc[4] = {};
    mma_loop4<Cc>(Ab, Wb, acc);
    int lane = threadIdx.x & 31;
    int mb = (lane >> 4) * 8;
#pragma unroll
    for (int t = 0; t < 4; ++t) {
        int n = nt4 * 64 + t * 16 + (lane & 15);
#pragma unroll
        for (int v = 0; v < 8; ++v) {
            size_t m = (size_t)mt * 16 + mb + v;
            float hv = acc[t][v] + fb[n];
            float g  = 0.5f * hv * (1.f + erff(hv * 0.70710678118f));
            H[m * 1024 + n] = f2bf(g);
        }
    }
}

// ---------------- kernel 10: fc2 GEMM + final residuals -> d_out ----------------
__global__ void gemm_fc2_kernel(const unsigned short* __restrict__ A,  // 16384 x 1024 bf16
                                const unsigned short* __restrict__ W,  // 256 x 1024 bf16
                                const float* __restrict__ fb,          // 256
                                const float* __restrict__ out_blc,     // 16384 x 256
                                const float* __restrict__ x,           // B,C,H,W
                                float* __restrict__ outp) {            // B,C,H,W
    int mt  = blockIdx.x * 4 + (threadIdx.x >> 5);
    int nt4 = blockIdx.y;
    const unsigned short* Ab = A + (size_t)mt * 16 * 1024;
    const unsigned short* Wb = W + (size_t)nt4 * 64 * 1024;
    v8f acc[4] = {};
    mma_loop4<1024>(Ab, Wb, acc);
    int lane = threadIdx.x & 31;
    int mb = (lane >> 4) * 8;
#pragma unroll
    for (int t = 0; t < 4; ++t) {
        int n = nt4 * 64 + t * 16 + (lane & 15);
#pragma unroll
        for (int v = 0; v < 8; ++v) {
            size_t m = (size_t)mt * 16 + mb + v;   // r = b*L + l
            int b = (int)(m >> 12), l = (int)(m & (LL - 1));
            float mval = acc[t][v] + fb[n];
            size_t xi = ((size_t)b * Cc + n) * LL + l;
            outp[xi] = out_blc[m * Cc + n] + mval + x[xi];
        }
    }
}

// ---------------- launcher ----------------
extern "C" void kernel_launch(void* const* d_in, const int* in_sizes, int n_in,
                              void* d_out, int out_size, void* d_ws, size_t ws_size,
                              hipStream_t stream) {
    const float* x      = (const float*)d_in[0];
    const float* ln_w   = (const float*)d_in[1];
    const float* ln_b   = (const float*)d_in[2];
    const float* skip   = (const float*)d_in[3];
    const float* proj_w = (const float*)d_in[4];
    const float* proj_b = (const float*)d_in[5];
    const float* in_w   = (const float*)d_in[6];
    const float* conv_w = (const float*)d_in[7];
    const float* conv_b = (const float*)d_in[8];
    const float* xproj_w= (const float*)d_in[9];
    const float* dt_w   = (const float*)d_in[10];
    const float* dt_b   = (const float*)d_in[11];
    const float* A_log  = (const float*)d_in[12];
    const float* Dp     = (const float*)d_in[13];
    const float* out_w  = (const float*)d_in[14];
    const float* fc1_w  = (const float*)d_in[15];
    const float* fc1_b  = (const float*)d_in[16];
    const float* fc2_w  = (const float*)d_in[17];
    const float* fc2_b  = (const float*)d_in[18];
    float* out = (float*)d_out;

    char* ws = (char*)d_ws;
    size_t o = 0;
    float*          chunks    = (float*)(ws + o);          o += MROWS * DM * 4;
    unsigned short* chunks_bf = (unsigned short*)(ws + o); o += MROWS * DM * 2;
    float*          xz        = (float*)(ws + o);          o += MROWS * 256 * 4;
    float*          xc        = (float*)(ws + o);          o += MROWS * DI * 4;
    float*          dtbuf     = (float*)(ws + o);          o += MROWS * DI * 4;
    float*          Bsb       = (float*)(ws + o);          o += MROWS * DS * 4;
    float*          Csb       = (float*)(ws + o);          o += MROWS * DS * 4;
    unsigned short* ybuf      = (unsigned short*)(ws + o); o += MROWS * DI * 2;
    float*          ym_blc    = (float*)(ws + o);          o += PROWS * Cc * 4;
    unsigned short* ym_ln     = (unsigned short*)(ws + o); o += PROWS * Cc * 2;
    float*          out_blc   = (float*)(ws + o);          o += PROWS * Cc * 4;
    unsigned short* out_bf    = (unsigned short*)(ws + o); o += PROWS * Cc * 2;
    unsigned short* hbuf      = (unsigned short*)(ws + o); o += PROWS * 1024 * 2;
    unsigned short* w_in_bf   = (unsigned short*)(ws + o); o += 256 * 64 * 2;
    unsigned short* w_out_bf  = (unsigned short*)(ws + o); o += 64 * 128 * 2;
    unsigned short* w_proj_bf = (unsigned short*)(ws + o); o += 256 * 256 * 2;
    unsigned short* w_fc1_bf  = (unsigned short*)(ws + o); o += 1024 * 256 * 2;
    unsigned short* w_fc2_bf  = (unsigned short*)(ws + o); o += 256 * 1024 * 2;
    float*          chP       = (float*)(ws + o);          o += (size_t)NSEQ * NCH * DI * DS * 4;
    float*          chS       = (float*)(ws + o);          o += (size_t)NSEQ * NCH * DI * DS * 4;
    float*          h0buf     = (float*)(ws + o);          o += (size_t)NSEQ * NCH * DI * DS * 4;
    (void)o; (void)ws_size; (void)n_in; (void)in_sizes; (void)out_size;

    // 0) convert weights to bf16 (tiny)
    cvt_bf16_kernel<<<dim3((256 * 64 + 255) / 256),   dim3(256), 0, stream>>>(in_w,   w_in_bf,   256 * 64);
    cvt_bf16_kernel<<<dim3((64 * 128 + 255) / 256),   dim3(256), 0, stream>>>(out_w,  w_out_bf,  64 * 128);
    cvt_bf16_kernel<<<dim3((256 * 256 + 255) / 256),  dim3(256), 0, stream>>>(proj_w, w_proj_bf, 256 * 256);
    cvt_bf16_kernel<<<dim3((1024 * 256 + 255) / 256), dim3(256), 0, stream>>>(fc1_w,  w_fc1_bf,  1024 * 256);
    cvt_bf16_kernel<<<dim3((256 * 1024 + 255) / 256), dim3(256), 0, stream>>>(fc2_w,  w_fc2_bf,  256 * 1024);

    // 1) LN1 + chunk (fp32 + bf16)
    ln1_chunks_kernel<<<dim3(Bb * LL), dim3(Cc), 0, stream>>>(x, ln_w, ln_b, chunks, chunks_bf);
    // 2) in-proj GEMM (65536 x 256 x 64)
    gemm_in_kernel<<<dim3(MROWS / 64, 256 / 64), dim3(128), 0, stream>>>(chunks_bf, w_in_bf, xz);
    // 3) conv + silu
    conv_silu_kernel<<<dim3((MROWS * DI) / 256), dim3(256), 0, stream>>>(xz, conv_w, conv_b, xc);
    // 4) x-proj + dt
    xproj_dt_kernel<<<dim3((unsigned)MROWS), dim3(DI), 0, stream>>>(xc, xproj_w, dt_w, dt_b,
                                                                    dtbuf, Bsb, Csb);
    // 5) chunked parallel scan: 256-wide parallel chunks, 16-step prefix, recompute
    scan_chunk1_kernel<<<dim3(NSEQ * NCH), dim3(DI), 0, stream>>>(dtbuf, Bsb, xc, A_log,
                                                                  chP, chS);
    scan_prefix_kernel<<<dim3((NSEQ * DI * DS) / 256), dim3(256), 0, stream>>>(chP, chS, h0buf);
    scan_chunk2_kernel<<<dim3(NSEQ * NCH), dim3(DI), 0, stream>>>(dtbuf, Bsb, Csb, xc, xz,
                                                                  A_log, Dp, h0buf, ybuf);
    // 6) out-proj GEMM + skip + scatter to (B,L,C)
    gemm_out_kernel<<<dim3(MROWS / 64, 1), dim3(128), 0, stream>>>(ybuf, w_out_bf, chunks,
                                                                   skip, ym_blc);
    // 7) LN2 -> bf16
    ln2_kernel<<<dim3((unsigned)PROWS), dim3(Cc), 0, stream>>>(ym_blc, ln_w, ln_b, ym_ln);
    // 8) proj GEMM + residual x (fp32 + bf16 outputs)
    gemm_proj_kernel<<<dim3(PROWS / 64, 256 / 64), dim3(128), 0, stream>>>(ym_ln, w_proj_bf,
                                                                           proj_b, x, out_blc, out_bf);
    // 9) fc1 + GELU -> bf16
    gemm_fc1_kernel<<<dim3(PROWS / 64, 1024 / 64), dim3(128), 0, stream>>>(out_bf, w_fc1_bf,
                                                                           fc1_b, hbuf);
    // 10) fc2 + final residuals -> (B,C,H,W)
    gemm_fc2_kernel<<<dim3(PROWS / 64, 256 / 64), dim3(128), 0, stream>>>(hbuf, w_fc2_bf, fc2_b,
                                                                          out_blc, x, out);
}
